// MultiHeadSelfAttention_54107997995120
// MI455X (gfx1250) — matrix operand order repaired and checked by reading, not compiled, not run on hardware
//
#include <hip/hip_runtime.h>
#include <stdint.h>

#define D_   1024
#define H_   16
#define DH_  64
#define S_   2048
#define B_   2
#define M_   (B_*S_)   /* 4096 tokens total */

typedef uint16_t u16;
typedef uint32_t u32;
typedef uint64_t u64;
typedef __attribute__((ext_vector_type(16))) __bf16 v16bf;
typedef __attribute__((ext_vector_type(8)))  __bf16 v8bf;
typedef __attribute__((ext_vector_type(8)))  float  v8f;
typedef __attribute__((ext_vector_type(4)))  u32    u32x4;

// ---- helpers -------------------------------------------------------------
__device__ __forceinline__ u16 f2bf(float f) {
  u32 x = __float_as_uint(f);
  return (u16)((x + 0x7fffu + ((x >> 16) & 1u)) >> 16);   // RNE
}
__device__ __forceinline__ u32 pack2(float lo, float hi) {
  return (u32)f2bf(lo) | ((u32)f2bf(hi) << 16);
}
__device__ __forceinline__ v16bf frag16(const u16* p0, const u16* p1) {
  union { v16bf v; v8bf h[2]; } u;
  u.h[0] = *reinterpret_cast<const v8bf*>(p0);
  u.h[1] = *reinterpret_cast<const v8bf*>(p1);
  return u.v;
}
__device__ __forceinline__ v8f wmma_bf16(v16bf a, v16bf b, v8f c) {
  return __builtin_amdgcn_wmma_f32_16x16x32_bf16(false, a, false, b,
                                                 (short)0, c, false, false);
}
// async 16B/lane Global->LDS copy; IOFFSET is added to BOTH lds and mem addr
#define ASYNC_CP16(lds, g, off_)                                        \
  asm volatile("global_load_async_to_lds_b128 %0, %1, off offset:" #off_ \
               :: "v"(lds), "v"(g) : "memory")
__device__ __forceinline__ void wait_async0() {
  asm volatile("s_wait_asynccnt 0x0" ::: "memory");
}
__device__ __forceinline__ u32 lds_addr(const void* p) {
  return (u32)(uintptr_t)p;   // low 32 bits of generic LDS ptr = DS address
}

// ---- kernel 0: fp32 -> bf16 activation conversion ------------------------
__global__ __launch_bounds__(256) void k_x2b(
    const float* __restrict__ qx, const float* __restrict__ kx,
    const float* __restrict__ vx,
    u16* __restrict__ Xq, u16* __restrict__ Xk, u16* __restrict__ Xv)
{
  const float* src = (blockIdx.z == 0) ? qx : (blockIdx.z == 1) ? kx : vx;
  u16*         dst = (blockIdx.z == 0) ? Xq : (blockIdx.z == 1) ? Xk : Xv;
  const size_t i = ((size_t)blockIdx.x * 256 + threadIdx.x) * 8;
  const float4* s = reinterpret_cast<const float4*>(src + i);
  float4 a = s[0], b = s[1];
  u32x4 o;
  o.x = pack2(a.x, a.y); o.y = pack2(a.z, a.w);
  o.z = pack2(b.x, b.y); o.w = pack2(b.z, b.w);
  *reinterpret_cast<u32x4*>(dst + i) = o;
}

// ---- kernel 1: weight transpose + fp32->bf16 -----------------------------
__global__ __launch_bounds__(256) void k_wt(
    const float* __restrict__ Wq, const float* __restrict__ Wk,
    const float* __restrict__ Wv, const float* __restrict__ Wo,
    u16* __restrict__ WqT, u16* __restrict__ WkT,
    u16* __restrict__ WvT, u16* __restrict__ WoT)
{
  const float* W  = (blockIdx.z == 0) ? Wq : (blockIdx.z == 1) ? Wk
                   : (blockIdx.z == 2) ? Wv : Wo;
  u16*         WT = (blockIdx.z == 0) ? WqT : (blockIdx.z == 1) ? WkT
                   : (blockIdx.z == 2) ? WvT : WoT;
  __shared__ float tile[32][33];
  const int tx = threadIdx.x, ty = threadIdx.y;
  const int k0 = blockIdx.y * 32, n0 = blockIdx.x * 32;
#pragma unroll
  for (int j = 0; j < 32; j += 8)
    tile[ty + j][tx] = W[(size_t)(k0 + ty + j) * D_ + n0 + tx];
  __syncthreads();
#pragma unroll
  for (int j = 0; j < 32; j += 8)
    WT[(size_t)(n0 + ty + j) * D_ + k0 + tx] = f2bf(tile[tx][ty + j]);
}

// ---- shared GEMM staging: 64xK A tile + 256xK B tile, all bf16 -----------
// 8 waves: wave stages 8 A-rows (1 async op) and 32 B-rows (4 async ops).
__device__ __forceinline__ void gemm_stage(
    u32 ldsA, u32 ldsB, const u16* __restrict__ Xb, const u16* __restrict__ WT,
    int m0, int n0, int k, int wave, int lane)
{
  const int arow = wave * 8 + (lane >> 2), acol = (lane & 3) * 8;
  const u32 lA = ldsA + (u32)((arow * 32 + acol) * 2);
  const u64 gA = (u64)(uintptr_t)(Xb + (size_t)(m0 + arow) * D_ + k + acol);
  ASYNC_CP16(lA, gA, 0);
  const int brow = wave * 32 + lane;
  const u32 lB = ldsB + (u32)(brow * 64);
  const u64 gB = (u64)(uintptr_t)(WT + (size_t)(n0 + brow) * D_ + k);
  ASYNC_CP16(lB, gB, 0);
  ASYNC_CP16(lB, gB, 16);
  ASYNC_CP16(lB, gB, 32);
  ASYNC_CP16(lB, gB, 48);
}

// ---- kernel 2: fused QKV projections -------------------------------------
// block 256 thr = 8 waves (2M x 4N); block tile 64M x 256N; wave tile 32x64.
__global__ __launch_bounds__(256) void k_qkv(
    const u16* __restrict__ Xqb, const u16* __restrict__ Xkb,
    const u16* __restrict__ Xvb,
    const u16* __restrict__ WqT, const u16* __restrict__ WkT,
    const u16* __restrict__ WvT,
    const float* __restrict__ bq, const float* __restrict__ bk,
    const float* __restrict__ bv,
    u16* __restrict__ Qb, u16* __restrict__ Kb, u16* __restrict__ Vt)
{
  const int proj = blockIdx.z;
  const u16*  Xb   = (proj == 0) ? Xqb : (proj == 1) ? Xkb : Xvb;
  const u16*  WT   = (proj == 0) ? WqT : (proj == 1) ? WkT : WvT;
  const float* bias = (proj == 0) ? bq : (proj == 1) ? bk : bv;

  __shared__ __align__(16) u16 As[2][64 * 32];
  __shared__ __align__(16) u16 Bs[2][256 * 32];

  const int tid  = threadIdx.x;
  const int m0   = blockIdx.y * 64;
  const int n0   = blockIdx.x * 256;
  const int wave = tid >> 5, lane = tid & 31;
  const int wm = wave >> 2, wn = wave & 3;
  const int nl = lane & 15, hl = lane >> 4;

  v8f acc[2][4] = {};

  gemm_stage(lds_addr(&As[0][0]), lds_addr(&Bs[0][0]), Xb, WT, m0, n0, 0, wave, lane);
  wait_async0();
  __syncthreads();

  for (int k0 = 0; k0 < D_; k0 += 32) {
    const int buf = (k0 >> 5) & 1;
    if (k0 + 32 < D_)
      gemm_stage(lds_addr(&As[buf ^ 1][0]), lds_addr(&Bs[buf ^ 1][0]),
                 Xb, WT, m0, n0, k0 + 32, wave, lane);

    const u16* a0 = &As[buf][(size_t)(wm * 32 + nl) * 32];
    const u16* a1 = &As[buf][(size_t)(wm * 32 + 16 + nl) * 32];
    v16bf af0 = frag16(a0 + hl * 8, a0 + 16 + hl * 8);
    v16bf af1 = frag16(a1 + hl * 8, a1 + 16 + hl * 8);
#pragma unroll
    for (int nt = 0; nt < 4; ++nt) {
      const u16* bbase = &Bs[buf][(size_t)(wn * 64 + nt * 16 + nl) * 32];
      v16bf bf = frag16(bbase + hl * 8, bbase + 16 + hl * 8);
      acc[0][nt] = wmma_bf16(af0, bf, acc[0][nt]);
      acc[1][nt] = wmma_bf16(af1, bf, acc[1][nt]);
    }
    wait_async0();
    __syncthreads();
  }

#pragma unroll
  for (int ms = 0; ms < 2; ++ms)
#pragma unroll
    for (int nt = 0; nt < 4; ++nt) {
      const int n  = n0 + wn * 64 + nt * 16 + nl;
      const float bb = bias[n];
      const int hh = n >> 6, dh = n & 63;
#pragma unroll
      for (int r = 0; r < 8; ++r) {
        const int t = m0 + wm * 32 + ms * 16 + r + hl * 8;
        const int b = t >> 11, s = t & 2047;
        const u16 val = f2bf(acc[ms][nt][r] + bb);
        if (proj == 2)
          Vt[(((size_t)(b * 16 + hh)) * 64 + dh) * 2048 + s] = val;   // [B,H,DH,S]
        else if (proj == 0)
          Qb[(((size_t)(b * 16 + hh)) * 2048 + s) * 64 + dh] = val;   // [B,H,S,DH]
        else
          Kb[(((size_t)(b * 16 + hh)) * 2048 + s) * 64 + dh] = val;
      }
    }
}

// ---- kernel 3: flash attention -------------------------------------------
__global__ __launch_bounds__(128) void k_attn(
    const u16* __restrict__ Qb, const u16* __restrict__ Kb,
    const u16* __restrict__ Vt, u16* __restrict__ Ctx)
{
  const int bh = blockIdx.y;            // b*16+h
  const int b = bh >> 4, h = bh & 15;
  const int q0 = blockIdx.x * 64;
  const int tid = threadIdx.x;
  const int wave = tid >> 5, lane = tid & 31;
  const int nl = lane & 15, hl = lane >> 4;

  __shared__ __align__(16) u16 Ks[2][32 * 64];     // keys x dh
  __shared__ __align__(16) u16 Vs[2][64 * 32];     // dh x keys
  __shared__ __align__(16) u16 Ps[4][16 * 32];     // per-wave P tile

  // per-lane async-copy bases (wave stages 8 K-rows + 16 V-rows per tile)
  const u32 kOffL = (u32)((lane >> 2) * 128 + (lane & 3) * 32);
  const u32 vOffL = (u32)((lane >> 1) * 64  + (lane & 1) * 32);
  const u32 lK[2] = { lds_addr(&Ks[0][0]) + (u32)(wave * 8 * 128) + kOffL,
                      lds_addr(&Ks[1][0]) + (u32)(wave * 8 * 128) + kOffL };
  const u32 lV[2] = { lds_addr(&Vs[0][0]) + (u32)(wave * 16 * 64) + vOffL,
                      lds_addr(&Vs[1][0]) + (u32)(wave * 16 * 64) + vOffL };
  u64 gK = (u64)(uintptr_t)Kb + ((u64)bh * 2048 + wave * 8) * 128
           + (u64)((lane >> 2) * 128 + (lane & 3) * 32);
  u64 gV = (u64)(uintptr_t)Vt + (((u64)bh * 64 + wave * 16) * 2048) * 2
           + (u64)((lane >> 1) * 4096 + (lane & 1) * 32);

  // preload Q fragments (K=dh, two 32-chunks)
  const u16* qrow = Qb + (((size_t)bh) * 2048 + q0 + wave * 16 + nl) * 64;
  v16bf qf0 = frag16(qrow + hl * 8,      qrow + 16 + hl * 8);
  v16bf qf1 = frag16(qrow + 32 + hl * 8, qrow + 48 + hl * 8);

  v8f acc[4] = {};
  float mrun[8], lrun[8];
#pragma unroll
  for (int r = 0; r < 8; ++r) { mrun[r] = -1e30f; lrun[r] = 0.f; }

  // prefetch first tile
  ASYNC_CP16(lK[0], gK, 0); ASYNC_CP16(lK[0], gK, 16);
  ASYNC_CP16(lV[0], gV, 0); ASYNC_CP16(lV[0], gV, 16);
  gK += 4096; gV += 64;
  wait_async0();
  __syncthreads();

  for (int kb = 0; kb < 2048; kb += 32) {
    const int buf = (kb >> 5) & 1, nb = buf ^ 1;
    if (kb + 32 < 2048) {
      ASYNC_CP16(lK[nb], gK, 0); ASYNC_CP16(lK[nb], gK, 16);
      ASYNC_CP16(lV[nb], gV, 0); ASYNC_CP16(lV[nb], gV, 16);
      gK += 4096; gV += 64;
    }

    // scores: S = Q @ K^T   (N = keys, K = dh)
    v8f s0 = {}, s1 = {};
    {
      const u16* k0p = &Ks[buf][(size_t)nl * 64];
      const u16* k1p = &Ks[buf][(size_t)(16 + nl) * 64];
      s0 = wmma_bf16(qf0, frag16(k0p + hl * 8,      k0p + 16 + hl * 8),      s0);
      s1 = wmma_bf16(qf0, frag16(k1p + hl * 8,      k1p + 16 + hl * 8),      s1);
      s0 = wmma_bf16(qf1, frag16(k0p + 32 + hl * 8, k0p + 48 + hl * 8), s0);
      s1 = wmma_bf16(qf1, frag16(k1p + 32 + hl * 8, k1p + 48 + hl * 8), s1);
    }

    // online softmax (row = r + 8*hl; reduce across 16 lanes in the half)
#pragma unroll
    for (int r = 0; r < 8; ++r) {
      s0[r] *= 0.125f; s1[r] *= 0.125f;               // 1/sqrt(64)
      float mx = fmaxf(s0[r], s1[r]);
      mx = fmaxf(mx, __shfl_xor(mx, 1, 32));
      mx = fmaxf(mx, __shfl_xor(mx, 2, 32));
      mx = fmaxf(mx, __shfl_xor(mx, 4, 32));
      mx = fmaxf(mx, __shfl_xor(mx, 8, 32));
      const float mi = fmaxf(mrun[r], mx);
      const float corr = __expf(mrun[r] - mi);
      mrun[r] = mi;
      const float p0 = __expf(s0[r] - mi);
      const float p1 = __expf(s1[r] - mi);
      s0[r] = p0; s1[r] = p1;
      float rs = p0 + p1;
      rs += __shfl_xor(rs, 1, 32);
      rs += __shfl_xor(rs, 2, 32);
      rs += __shfl_xor(rs, 4, 32);
      rs += __shfl_xor(rs, 8, 32);
      lrun[r] = lrun[r] * corr + rs;
      acc[0][r] *= corr; acc[1][r] *= corr; acc[2][r] *= corr; acc[3][r] *= corr;
    }

    // P (C-layout) -> LDS -> A-layout fragment, intra-wave only
    u16* prow = &Ps[wave][0];
#pragma unroll
    for (int r = 0; r < 8; ++r) {
      const int m = r + hl * 8;
      prow[m * 32 + nl]      = f2bf(s0[r]);
      prow[m * 32 + 16 + nl] = f2bf(s1[r]);
    }
    asm volatile("s_wait_dscnt 0x0" ::: "memory");
    const u16* pbase = prow + (size_t)nl * 32;
    v16bf pf = frag16(pbase + hl * 8, pbase + 16 + hl * 8);

    // ctx += P @ V   (N = dh, K = keys; Vs rows are dh over keys)
#pragma unroll
    for (int nt = 0; nt < 4; ++nt) {
      const u16* vb = &Vs[buf][(size_t)(nt * 16 + nl) * 32];
      acc[nt] = wmma_bf16(pf, frag16(vb + hl * 8, vb + 16 + hl * 8), acc[nt]);
    }
    wait_async0();
    __syncthreads();
  }

  // normalize and write ctx back into [B,S,D] layout (bf16)
  const size_t trow = (size_t)b * 2048 + q0 + wave * 16;
#pragma unroll
  for (int r = 0; r < 8; ++r) {
    const float inv = 1.0f / lrun[r];
    const int m = r + hl * 8;
    const size_t base = (trow + m) * D_ + h * 64;
#pragma unroll
    for (int nt = 0; nt < 4; ++nt)
      Ctx[base + nt * 16 + nl] = f2bf(acc[nt][r] * inv);
  }
}

// ---- kernel 4: output projection (bf16 A/B, fp32 out) --------------------
__global__ __launch_bounds__(256) void k_oproj(
    const u16* __restrict__ Ctx, const u16* __restrict__ WoT,
    const float* __restrict__ bo, float* __restrict__ out)
{
  __shared__ __align__(16) u16 As[2][64 * 32];
  __shared__ __align__(16) u16 Bs[2][256 * 32];

  const int tid  = threadIdx.x;
  const int m0   = blockIdx.y * 64;
  const int n0   = blockIdx.x * 256;
  const int wave = tid >> 5, lane = tid & 31;
  const int wm = wave >> 2, wn = wave & 3;
  const int nl = lane & 15, hl = lane >> 4;

  v8f acc[2][4] = {};

  gemm_stage(lds_addr(&As[0][0]), lds_addr(&Bs[0][0]), Ctx, WoT, m0, n0, 0, wave, lane);
  wait_async0();
  __syncthreads();

  for (int k0 = 0; k0 < D_; k0 += 32) {
    const int buf = (k0 >> 5) & 1;
    if (k0 + 32 < D_)
      gemm_stage(lds_addr(&As[buf ^ 1][0]), lds_addr(&Bs[buf ^ 1][0]),
                 Ctx, WoT, m0, n0, k0 + 32, wave, lane);

    const u16* a0 = &As[buf][(size_t)(wm * 32 + nl) * 32];
    const u16* a1 = &As[buf][(size_t)(wm * 32 + 16 + nl) * 32];
    v16bf af0 = frag16(a0 + hl * 8, a0 + 16 + hl * 8);
    v16bf af1 = frag16(a1 + hl * 8, a1 + 16 + hl * 8);
#pragma unroll
    for (int nt = 0; nt < 4; ++nt) {
      const u16* bbase = &Bs[buf][(size_t)(wn * 64 + nt * 16 + nl) * 32];
      v16bf bf = frag16(bbase + hl * 8, bbase + 16 + hl * 8);
      acc[0][nt] = wmma_bf16(af0, bf, acc[0][nt]);
      acc[1][nt] = wmma_bf16(af1, bf, acc[1][nt]);
    }
    wait_async0();
    __syncthreads();
  }

#pragma unroll
  for (int ms = 0; ms < 2; ++ms)
#pragma unroll
    for (int nt = 0; nt < 4; ++nt) {
      const int n = n0 + wn * 64 + nt * 16 + nl;
      const float bb = bo[n];
#pragma unroll
      for (int r = 0; r < 8; ++r)
        out[(size_t)(m0 + wm * 32 + ms * 16 + r + hl * 8) * D_ + n] =
            acc[ms][nt][r] + bb;
    }
}

// ---- launcher ------------------------------------------------------------
extern "C" void kernel_launch(void* const* d_in, const int* in_sizes, int n_in,
                              void* d_out, int out_size, void* d_ws, size_t ws_size,
                              hipStream_t stream) {
  (void)in_sizes; (void)n_in; (void)out_size; (void)ws_size;
  const float* qx = (const float*)d_in[0];
  const float* kx = (const float*)d_in[1];
  const float* vx = (const float*)d_in[2];
  const float* Wq = (const float*)d_in[3];
  const float* bq = (const float*)d_in[4];
  const float* Wk = (const float*)d_in[5];
  const float* bk = (const float*)d_in[6];
  const float* Wv = (const float*)d_in[7];
  const float* bv = (const float*)d_in[8];
  const float* Wo = (const float*)d_in[9];
  const float* bo = (const float*)d_in[10];
  float* out = (float*)d_out;

  char* ws = (char*)d_ws;                    // 64 MB used
  u16* WqT = (u16*)(ws + (0ull  << 20));
  u16* WkT = (u16*)(ws + (2ull  << 20));
  u16* WvT = (u16*)(ws + (4ull  << 20));
  u16* WoT = (u16*)(ws + (6ull  << 20));
  u16* Qb  = (u16*)(ws + (8ull  << 20));
  u16* Kb  = (u16*)(ws + (16ull << 20));
  u16* Vt  = (u16*)(ws + (24ull << 20));
  u16* Ctx = (u16*)(ws + (32ull << 20));
  u16* Xqb = (u16*)(ws + (40ull << 20));
  u16* Xkb = (u16*)(ws + (48ull << 20));
  u16* Xvb = (u16*)(ws + (56ull << 20));

  k_x2b  <<<dim3((M_ * D_) / (256 * 8), 1, 3), dim3(256, 1, 1), 0, stream>>>(
      qx, kx, vx, Xqb, Xkb, Xvb);
  k_wt   <<<dim3(D_ / 32, D_ / 32, 4), dim3(32, 8, 1), 0, stream>>>(
      Wq, Wk, Wv, Wo, WqT, WkT, WvT, WoT);
  k_qkv  <<<dim3(D_ / 256, M_ / 64, 3), dim3(256, 1, 1), 0, stream>>>(
      Xqb, Xkb, Xvb, WqT, WkT, WvT, bq, bk, bv, Qb, Kb, Vt);
  k_attn <<<dim3(S_ / 64, B_ * H_, 1), dim3(128, 1, 1), 0, stream>>>(
      Qb, Kb, Vt, Ctx);
  k_oproj<<<dim3(D_ / 256, M_ / 64, 1), dim3(256, 1, 1), 0, stream>>>(
      Ctx, WoT, bo, out);
}
